// MKAForGPT2Attention_66340064854742
// MI455X (gfx1250) — matrix operand order repaired and checked by reading, not compiled, not run on hardware
//
#include <hip/hip_runtime.h>
#include <math.h>

// MKA attention for GPT2 on gfx1250 (MI455X), wave32 + WMMA f16->f32 +
// async global->LDS staging. Levels 2 (mean-broadcast) and 3 (zeros)
// degenerate to per-batch / global constant vectors; only level 1 needs
// real attention.

namespace {
constexpr int kB = 2, kT = 2048, kC = 1024, kH = 16, kD = 64;
constexpr int kM = kB * kT;  // 4096 token rows
constexpr float kScale = 0.125f;  // D^-0.5, folded into Q

constexpr int kModeF16 = 0;       // f16 row-major store
constexpr int kModeF16Scale = 1;  // f16 row-major, *SCALE (Q)
constexpr int kModeVT = 2;        // f16 transposed per-head [b][h][d][t] (V)
constexpr int kModeTanh = 3;      // tanh + f16 row-major (routing hidden)
constexpr int kModeOut = 4;       // f32 routed-combine epilogue (output proj)
}  // namespace

typedef __attribute__((ext_vector_type(16))) _Float16 v16h;
typedef __attribute__((ext_vector_type(8)))  _Float16 v8h;
typedef __attribute__((ext_vector_type(8)))  float    v8f;

__device__ __forceinline__ v8f wmma16x16x32(v16h a, v16h b, v8f c) {
  // D = A(16x32 f16) x B(32x16 f16) + C(16x16 f32)
  return __builtin_amdgcn_wmma_f32_16x16x32_f16(
      /*neg_a=*/false, a, /*neg_b=*/false, b, /*c_mod=*/(short)0, c,
      /*reuse_a=*/false, /*reuse_b=*/false);
}

// A-fragment, 16x32 f16: lane (m = lane&15, g = lane>>4) holds
// K = g*16..g*16+15 contiguous in elements 0..15.
__device__ __forceinline__ v16h load_a_frag(const _Float16* p) {
  v8h lo = *(const v8h*)p;
  v8h hi = *(const v8h*)(p + 8);
  v16h r;
#pragma unroll
  for (int i = 0; i < 8; ++i) { r[i] = lo[i]; r[i + 8] = hi[i]; }
  return r;
}
// B-fragment, 32x16 f16: lane (n = lane&15, g = lane>>4) holds K = g*16..+15
// contiguous -> a single 32-byte v16h load from a K-contiguous buffer.

// Async copy of 16 bytes: global (per-lane addr) -> LDS (per-lane offset).
// Tracked by ASYNCcnt; low 32 bits of a flat shared pointer are the LDS offset.
__device__ __forceinline__ void async_copy_b128(const void* gsrc, void* lds_dst) {
  const unsigned loff = (unsigned)(unsigned long long)lds_dst;
  asm volatile("global_load_async_to_lds_b128 %0, %1, off"
               :: "v"(loff), "v"((unsigned long long)gsrc) : "memory");
}
__device__ __forceinline__ void wait_async_zero() {
  asm volatile("s_wait_asynccnt 0x0" ::: "memory");
}

// ---------------------------------------------------------------- utilities
__global__ void cvt_f32_to_f16(const float* __restrict__ in,
                               _Float16* __restrict__ out, int n) {
  int i = blockIdx.x * blockDim.x + threadIdx.x;
  const int stride = gridDim.x * blockDim.x;
  for (; i < n; i += stride) out[i] = (_Float16)in[i];
}

__global__ void transpose_to_f16(const float* __restrict__ w,
                                 _Float16* __restrict__ wt, int k, int n) {
  // wt[nn][kk] = w[kk][nn]  (weights become [N][K], K-contiguous)
  const int idx = blockIdx.x * blockDim.x + threadIdx.x;
  if (idx >= k * n) return;
  const int nn = idx / k, kk = idx % k;
  wt[(size_t)nn * k + kk] = (_Float16)w[(size_t)kk * n + nn];
}

__global__ void col_mean(const float* __restrict__ x, float* __restrict__ mean) {
  const int idx = blockIdx.x * blockDim.x + threadIdx.x;  // over B*C
  if (idx >= kB * kC) return;
  const int b = idx / kC, c = idx % kC;
  const float* p = x + (size_t)b * kT * kC + c;
  float s = 0.f;
  for (int t = 0; t < kT; ++t) s += p[(size_t)t * kC];
  mean[idx] = s * (1.0f / (float)kT);
}

__global__ void gemv_f32(const float* __restrict__ in, const float* __restrict__ w,
                         const float* __restrict__ bias, float* __restrict__ out,
                         int rows, int k, int n) {
  const int j = blockIdx.x * blockDim.x + threadIdx.x;
  if (j >= n) return;
  for (int r = 0; r < rows; ++r) {
    float s = bias[j];
    const float* inr = in + (size_t)r * k;
    for (int kk = 0; kk < k; ++kk) s += inr[kk] * w[(size_t)kk * n + j];
    out[(size_t)r * n + j] = s;
  }
}

__global__ void copy_f32(const float* __restrict__ src, float* __restrict__ dst,
                         int n) {
  const int i = blockIdx.x * blockDim.x + threadIdx.x;
  if (i < n) dst[i] = src[i];
}

// ---------------------------------------------------------------- WMMA GEMM
// One wave computes a 32(M) x 64(N) tile: 2 A-frags x 4 B-frags -> 8 WMMAs
// per 32-deep K step (each B-frag reused twice). A [M][K] f16, Bt [N][K] f16.
__global__ void __launch_bounds__(128)
gemm_wmma_f16(const _Float16* __restrict__ A, const _Float16* __restrict__ Bt,
              const float* __restrict__ bias, void* __restrict__ out,
              int M, int N, int K, int mode,
              const float* __restrict__ routing,  // [M][3]   (kModeOut)
              const float* __restrict__ lv2,      // [B][N]   (kModeOut)
              const float* __restrict__ lv3) {    // [N]      (kModeOut)
  const int tiles_n = N >> 6;
  const int wid = blockIdx.x * (blockDim.x >> 5) + (threadIdx.x >> 5);
  if (wid >= (M >> 5) * tiles_n) return;
  const int m0 = (wid / tiles_n) << 5;
  const int n0 = (wid % tiles_n) << 6;
  const int lane = threadIdx.x & 31;
  const int lm = lane & 15, g = lane >> 4;

  const _Float16* arow0 = A + (size_t)(m0 + lm) * K + g * 16;
  const _Float16* arow1 = arow0 + (size_t)16 * K;
  const _Float16* brow = Bt + (size_t)(n0 + lm) * K + g * 16;
  v8f acc[2][4] = {};
  for (int k0 = 0; k0 < K; k0 += 32) {
    __builtin_prefetch(arow0 + k0 + 256, 0, 1);
    const v16h af0 = load_a_frag(arow0 + k0);
    const v16h af1 = load_a_frag(arow1 + k0);
#pragma unroll
    for (int t = 0; t < 4; ++t) {
      const v16h bf = *(const v16h*)(brow + (size_t)(t * 16) * K + k0);
      acc[0][t] = wmma16x16x32(af0, bf, acc[0][t]);
      acc[1][t] = wmma16x16x32(af1, bf, acc[1][t]);
    }
  }

#pragma unroll
  for (int mh = 0; mh < 2; ++mh) {
    const int mb = m0 + mh * 16;
    if (mode == kModeVT) {
      // scatter into per-head transposed V: vt[b][h][d][t], t contiguous
      _Float16* vt = (_Float16*)out;
      const int b = mb / kT;
      const int tbase = (mb - b * kT) + g * 8;
#pragma unroll
      for (int t = 0; t < 4; ++t) {
        const int col = n0 + t * 16 + lm;
        const float bc = bias[col];
        const int h = col >> 6, d = col & 63;
        v8h pk;
#pragma unroll
        for (int r = 0; r < 8; ++r) pk[r] = (_Float16)(acc[mh][t][r] + bc);
        *(v8h*)(vt + ((size_t)(b * kH + h) * kD + d) * kT + tbase) = pk;
      }
    } else if (mode == kModeOut) {
      float* of = (float*)out;
      const int b = mb / kT;
#pragma unroll
      for (int r = 0; r < 8; ++r) {
        const int row = mb + r + 8 * g;
        const float r0 = routing[row * 3 + 0];
        const float r1 = routing[row * 3 + 1];
        const float r2 = routing[row * 3 + 2];
#pragma unroll
        for (int t = 0; t < 4; ++t) {
          const int col = n0 + t * 16 + lm;
          of[(size_t)row * N + col] =
              (acc[mh][t][r] + bias[col]) * r0 +
              r1 * lv2[(size_t)b * N + col] + r2 * lv3[col];
        }
      }
    } else {
      _Float16* oh = (_Float16*)out;
#pragma unroll
      for (int t = 0; t < 4; ++t) {
        const int col = n0 + t * 16 + lm;
        const float bc = bias[col];
#pragma unroll
        for (int r = 0; r < 8; ++r) {
          float v = acc[mh][t][r] + bc;
          if (mode == kModeF16Scale) v *= kScale;
          if (mode == kModeTanh) v = tanhf(v);
          oh[(size_t)(mb + r + 8 * g) * N + col] = (_Float16)v;
        }
      }
    }
  }
}

// ------------------------------------------------------- routing gate (Cx3)
__global__ void __launch_bounds__(128)
routing_softmax(const _Float16* __restrict__ rh, const float* __restrict__ wr2,
                const float* __restrict__ br2, float* __restrict__ routing) {
  const int wid = blockIdx.x * (blockDim.x >> 5) + (threadIdx.x >> 5);
  if (wid >= kM) return;
  const int lane = threadIdx.x & 31;
  const _Float16* row = rh + (size_t)wid * kC;
  float s0 = 0.f, s1 = 0.f, s2 = 0.f;
  for (int k = lane; k < kC; k += 32) {
    const float v = (float)row[k];
    s0 += v * wr2[k * 3 + 0];
    s1 += v * wr2[k * 3 + 1];
    s2 += v * wr2[k * 3 + 2];
  }
#pragma unroll
  for (int m = 16; m >= 1; m >>= 1) {
    s0 += __shfl_xor(s0, m, 32);
    s1 += __shfl_xor(s1, m, 32);
    s2 += __shfl_xor(s2, m, 32);
  }
  if (lane == 0) {
    s0 += br2[0]; s1 += br2[1]; s2 += br2[2];
    const float mx = fmaxf(s0, fmaxf(s1, s2));
    const float e0 = __expf(s0 - mx), e1 = __expf(s1 - mx), e2 = __expf(s2 - mx);
    const float inv = 1.0f / (e0 + e1 + e2);
    routing[wid * 3 + 0] = e0 * inv;
    routing[wid * 3 + 1] = e1 * inv;
    routing[wid * 3 + 2] = e2 * inv;
  }
}

// ------------------------------------------------------- flash attention L1
// One block = 4 waves = 4 consecutive 16-query tiles of one (b,h).
// K/V tiles for each 32-key step are staged into LDS once per block with
// async global->LDS copies (ASYNCcnt) and shared by all 4 waves.
__global__ void __launch_bounds__(128)
flash_attention(const _Float16* __restrict__ qh, const _Float16* __restrict__ kh,
                const _Float16* __restrict__ vt, _Float16* __restrict__ attn) {
  __shared__ __align__(32) _Float16 klds[32 * 64];      // [key][d]
  __shared__ __align__(32) _Float16 vlds[64 * 32];      // [d][key]
  __shared__ __align__(32) _Float16 plds[4][16 * 32];   // per-wave P tile
  const int tid = threadIdx.x;
  const int wave = tid >> 5, lane = tid & 31;
  const int lm = lane & 15, g = lane >> 4;
  constexpr int kBlkPerBH = kT / 64;  // 32 blocks per (b,h)
  const int bh = blockIdx.x / kBlkPerBH;
  const int h = bh % kH, b = bh / kH;
  const int q0 = (blockIdx.x % kBlkPerBH) * 64 + wave * 16;

  const _Float16* qp = qh + (size_t)(b * kT + q0 + lm) * kC + h * kD + g * 16;
  const v16h qf0 = load_a_frag(qp);        // D = 0..31  (SCALE pre-folded)
  const v16h qf1 = load_a_frag(qp + 32);   // D = 32..63

  const _Float16* kbase = kh + (size_t)b * kT * kC + h * kD;   // [t][d]
  const _Float16* vbase = vt + (size_t)(b * kH + h) * kD * kT; // [d][t]
  _Float16* pl = &plds[wave][0];

  float mrow[8], lrow[8];
#pragma unroll
  for (int r = 0; r < 8; ++r) { mrow[r] = -3.0e38f; lrow[r] = 0.f; }
  v8f o[4] = {};

  for (int k0 = 0; k0 < kT; k0 += 32) {
    __syncthreads();  // fence previous iteration's LDS reads vs new DMA writes
    // stage K (32x64 halves, 8x16B chunks/row) and V (64x32 halves, 4/row)
#pragma unroll
    for (int c = tid; c < 256; c += 128) {
      const int kr = c >> 3, ko = (c & 7) * 8;
      async_copy_b128(kbase + (size_t)(k0 + kr) * kC + ko, klds + kr * 64 + ko);
      const int vr = c >> 2, vo = (c & 3) * 8;
      async_copy_b128(vbase + (size_t)vr * kT + k0 + vo, vlds + vr * 32 + vo);
    }
    wait_async_zero();   // this wave's chunks are in LDS
    __syncthreads();     // all waves' chunks visible

    // S = Q K^T : 2 key n-tiles x 2 d-chunks, B-frags from LDS
    v8f s[2] = {};
#pragma unroll
    for (int nt = 0; nt < 2; ++nt) {
      const _Float16* kp = klds + (nt * 16 + lm) * 64 + g * 16;
      s[nt] = wmma16x16x32(qf0, *(const v16h*)kp, s[nt]);
      s[nt] = wmma16x16x32(qf1, *(const v16h*)(kp + 32), s[nt]);
    }
    // online softmax: C/D-layout row r lives in lane group g as row r+8g
#pragma unroll
    for (int r = 0; r < 8; ++r) {
      float mx = fmaxf(s[0][r], s[1][r]);
#pragma unroll
      for (int m = 8; m >= 1; m >>= 1) mx = fmaxf(mx, __shfl_xor(mx, m, 32));
      const float mnew = fmaxf(mrow[r], mx);
      const float alpha = __expf(mrow[r] - mnew);
      const float p0 = __expf(s[0][r] - mnew);
      const float p1 = __expf(s[1][r] - mnew);
      float ps = p0 + p1;
#pragma unroll
      for (int m = 8; m >= 1; m >>= 1) ps += __shfl_xor(ps, m, 32);
      lrow[r] = lrow[r] * alpha + ps;
      mrow[r] = mnew;
#pragma unroll
      for (int t = 0; t < 4; ++t) o[t][r] *= alpha;
      pl[(r + 8 * g) * 32 + lm] = (_Float16)p0;        // keys k0..k0+15
      pl[(r + 8 * g) * 32 + 16 + lm] = (_Float16)p1;   // keys k0+16..k0+31
    }
    // repack P (C/D layout) -> A-fragment via per-wave LDS (DS in-order)
    const v16h pf = load_a_frag(pl + lm * 32 + g * 16);
#pragma unroll
    for (int dt = 0; dt < 4; ++dt) {
      const _Float16* vp = vlds + (dt * 16 + lm) * 32 + g * 16;
      o[dt] = wmma16x16x32(pf, *(const v16h*)vp, o[dt]);
    }
  }

  _Float16* ob = attn + (size_t)(b * kT + q0) * kC + h * kD;
#pragma unroll
  for (int r = 0; r < 8; ++r) {
    const float inv = 1.0f / lrow[r];
#pragma unroll
    for (int dt = 0; dt < 4; ++dt)
      ob[(size_t)(r + 8 * g) * kC + dt * 16 + lm] = (_Float16)(o[dt][r] * inv);
  }
}

// ---------------------------------------------------------------- launcher
extern "C" void kernel_launch(void* const* d_in, const int* in_sizes, int n_in,
                              void* d_out, int out_size, void* d_ws, size_t ws_size,
                              hipStream_t stream) {
  const float* x   = (const float*)d_in[0];
  const float* Wq  = (const float*)d_in[1];
  const float* bq  = (const float*)d_in[2];
  const float* Wk  = (const float*)d_in[3];
  const float* bk  = (const float*)d_in[4];
  const float* Wv  = (const float*)d_in[5];
  const float* bv  = (const float*)d_in[6];
  const float* Wo  = (const float*)d_in[7];
  const float* bo  = (const float*)d_in[8];
  const float* Wr1 = (const float*)d_in[9];
  const float* br1 = (const float*)d_in[10];
  const float* Wr2 = (const float*)d_in[11];
  const float* br2 = (const float*)d_in[12];
  float* out = (float*)d_out;

  char* p = (char*)d_ws;
  auto take = [&](size_t bytes) {
    char* r = p;
    p += (bytes + 255) & ~(size_t)255;
    return r;
  };
  _Float16* xh   = (_Float16*)take((size_t)kM * kC * 2);
  _Float16* qhb  = (_Float16*)take((size_t)kM * kC * 2);
  _Float16* khb  = (_Float16*)take((size_t)kM * kC * 2);
  _Float16* vtb  = (_Float16*)take((size_t)kM * kC * 2);
  _Float16* rhb  = (_Float16*)take((size_t)kM * kC * 2);
  _Float16* aob  = (_Float16*)take((size_t)kM * kC * 2);
  _Float16* Wqt  = (_Float16*)take((size_t)kC * kC * 2);
  _Float16* Wkt  = (_Float16*)take((size_t)kC * kC * 2);
  _Float16* Wvt  = (_Float16*)take((size_t)kC * kC * 2);
  _Float16* Wot  = (_Float16*)take((size_t)kC * kC * 2);
  _Float16* Wr1t = (_Float16*)take((size_t)kC * kC * 2);
  float* meanb   = (float*)take((size_t)kB * kC * 4);
  float* tmp23   = (float*)take((size_t)3 * kC * 4);
  float* lv23    = (float*)take((size_t)3 * kC * 4);
  float* routing = (float*)take((size_t)kM * 3 * 4);

  // activations -> f16, batch means
  cvt_f32_to_f16<<<2048, 256, 0, stream>>>(x, xh, kM * kC);
  col_mean<<<(kB * kC + 255) / 256, 256, 0, stream>>>(x, meanb);

  // weights -> f16 [N][K]
  const int we = kC * kC, wb = (we + 255) / 256;
  transpose_to_f16<<<wb, 256, 0, stream>>>(Wq, Wqt, kC, kC);
  transpose_to_f16<<<wb, 256, 0, stream>>>(Wk, Wkt, kC, kC);
  transpose_to_f16<<<wb, 256, 0, stream>>>(Wv, Wvt, kC, kC);
  transpose_to_f16<<<wb, 256, 0, stream>>>(Wo, Wot, kC, kC);
  transpose_to_f16<<<wb, 256, 0, stream>>>(Wr1, Wr1t, kC, kC);

  // projections (WMMA), 32x64 tile per wave, 4 waves per block
  const int gblocks = ((kM / 32) * (kC / 64)) / 4;
  gemm_wmma_f16<<<gblocks, 128, 0, stream>>>(xh, Wqt, bq, qhb, kM, kC, kC,
                                             kModeF16Scale, nullptr, nullptr, nullptr);
  gemm_wmma_f16<<<gblocks, 128, 0, stream>>>(xh, Wkt, bk, khb, kM, kC, kC,
                                             kModeF16, nullptr, nullptr, nullptr);
  gemm_wmma_f16<<<gblocks, 128, 0, stream>>>(xh, Wvt, bv, vtb, kM, kC, kC,
                                             kModeVT, nullptr, nullptr, nullptr);
  gemm_wmma_f16<<<gblocks, 128, 0, stream>>>(xh, Wr1t, br1, rhb, kM, kC, kC,
                                             kModeTanh, nullptr, nullptr, nullptr);

  // routing gate
  routing_softmax<<<kM / 4, 128, 0, stream>>>(rhb, Wr2, br2, routing);

  // degenerate levels: tmp23 = [mean@Wv+bv (x2); bv]; lv23 = tmp23@Wo+bo
  gemv_f32<<<(kC + 255) / 256, 256, 0, stream>>>(meanb, Wv, bv, tmp23, kB, kC, kC);
  copy_f32<<<(kC + 255) / 256, 256, 0, stream>>>(bv, tmp23 + 2 * kC, kC);
  gemv_f32<<<(kC + 255) / 256, 256, 0, stream>>>(tmp23, Wo, bo, lv23, 3, kC, kC);

  // level-1 attention: one block per 64 queries of one (b,h)
  flash_attention<<<kB * kH * (kT / 64), 128, 0, stream>>>(qhb, khb, vtb, aob);

  // output projection + routed combine (WMMA)
  gemm_wmma_f16<<<gblocks, 128, 0, stream>>>(aob, Wot, bo, out, kM, kC, kC,
                                             kModeOut, routing, lv23, lv23 + 2 * kC);
}
